// TPALSTM_10144712753754
// MI455X (gfx1250) — matrix operand order repaired and checked by reading, not compiled, not run on hardware
//
#include <hip/hip_runtime.h>
#include <hip/hip_bf16.h>
#include <math.h>

// ---------------------------------------------------------------------------
// TPA-LSTM on gfx1250 (MI455X).  All GEMMs via v_wmma_f32_16x16x32_f16.
// B=256, T=256, FEAT=256, HEADS=8, hd=32, HIDDEN=512, N_LAYERS=2.
// Round 4: 6-arg tensor_load_to_lds (clang-23 toolchain arity) for the
// A-tile staging in the hot sequential lstm_gates kernel.
// ---------------------------------------------------------------------------

#define BB    256
#define TT    256
#define FEATD 256
#define NHEAD 8
#define HD    32
#define HID   512
#define G4    2048   // 4*HID

typedef _Float16 v16h __attribute__((ext_vector_type(16)));
typedef _Float16 v8h  __attribute__((ext_vector_type(8)));
typedef float    v8f  __attribute__((ext_vector_type(8)));
typedef float    v4f  __attribute__((ext_vector_type(4)));
typedef unsigned int u32x4 __attribute__((ext_vector_type(4)));
typedef int          i32x8 __attribute__((ext_vector_type(8)));
typedef int          i32x4 __attribute__((ext_vector_type(4)));

__device__ __forceinline__ int laneid() { return threadIdx.x & 31; }

// ---- WMMA operand loaders (layouts per CDNA5 ISA 7.12.2, wave32) ----------

// A-matrix 16x32 f16: lane l<16 holds row l, K in {0..7, 16..23};
// lane l>=16 holds row l-16, K in {8..15, 24..31}.
__device__ __forceinline__ v16h ldA16(const _Float16* tile, int ld) {
  const int l  = laneid();
  const _Float16* p = tile + (l & 15) * ld + ((l & 16) ? 8 : 0);
  v8h lo = *(const v8h*)p;
  v8h hi = *(const v8h*)(p + 16);
  v16h a;
#pragma unroll
  for (int i = 0; i < 8; ++i) { a[i] = lo[i]; a[8 + i] = hi[i]; }
  return a;
}

// Same A layout, sourced from f32 with convert.
__device__ __forceinline__ v16h ldA32cvt(const float* tile, int ld) {
  const int l = laneid();
  const float* p = tile + (l & 15) * ld + ((l & 16) ? 8 : 0);
  v4f a0 = *(const v4f*)p;
  v4f a1 = *(const v4f*)(p + 4);
  v4f b0 = *(const v4f*)(p + 16);
  v4f b1 = *(const v4f*)(p + 20);
  v16h a;
#pragma unroll
  for (int i = 0; i < 4; ++i) {
    a[i]      = (_Float16)a0[i];
    a[4 + i]  = (_Float16)a1[i];
    a[8 + i]  = (_Float16)b0[i];
    a[12 + i] = (_Float16)b1[i];
  }
  return a;
}

// B-matrix 32x16 f16 from a source laid out [col][k] (contiguous along K).
__device__ __forceinline__ v16h ldB16(const _Float16* tile, int ld) {
  const int l = laneid();
  const _Float16* p = tile + (l & 15) * ld + ((l & 16) ? 8 : 0);
  v8h lo = *(const v8h*)p;
  v8h hi = *(const v8h*)(p + 16);
  v16h b;
#pragma unroll
  for (int i = 0; i < 8; ++i) { b[i] = lo[i]; b[8 + i] = hi[i]; }
  return b;
}

// ---- Coalesced C-tile stores via per-wave LDS staging ---------------------
__device__ __forceinline__ void stC16co(_Float16* gtile, int ld, v8f c,
                                        _Float16* lds /*256 halfs*/) {
  const int l = laneid(); const int col = l & 15; const int rb = (l & 16) ? 8 : 0;
#pragma unroll
  for (int v = 0; v < 8; ++v) lds[(rb + v) * 16 + col] = (_Float16)c[v];
  v8h r = *(const v8h*)(lds + l * 8);
  *(v8h*)(gtile + (l >> 1) * ld + (l & 1) * 8) = r;
}
__device__ __forceinline__ void stC32co(float* gtile, int ld, v8f c,
                                        float* lds /*256 floats*/) {
  const int l = laneid(); const int col = l & 15; const int rb = (l & 16) ? 8 : 0;
#pragma unroll
  for (int v = 0; v < 8; ++v) lds[(rb + v) * 16 + col] = c[v];
  v8f r = *(const v8f*)(lds + l * 8);
  *(v8f*)(gtile + (l >> 1) * ld + (l & 1) * 8) = r;
}
__device__ __forceinline__ void stCtr16co(_Float16* gtile, int ld, v8f c,
                                          _Float16* lds /*256 halfs*/) {
  const int l = laneid(); const int col = l & 15; const int rb = (l & 16) ? 8 : 0;
  v8h pk;
#pragma unroll
  for (int v = 0; v < 8; ++v) pk[v] = (_Float16)c[v];
  *(v8h*)(lds + col * 16 + rb) = pk;          // (r,c) at lds[c*16 + r]
  v8h r = *(const v8h*)(lds + l * 8);
  *(v8h*)(gtile + (l >> 1) * ld + (l & 1) * 8) = r;
}

#define WMMA(a, b, c) __builtin_amdgcn_wmma_f32_16x16x32_f16(false, (a), false, (b), (short)0, (c), false, false)

// ---- Tensor Data Mover: 2D tile global -> LDS (D# per CDNA5 ISA ch.8) -----
#if __has_builtin(__builtin_amdgcn_tensor_load_to_lds)
#define HAVE_TDM 1
// widthUnits/strideUnits in 8-byte units (data_size=3); rows = tile_dim1.
__device__ __forceinline__ void tdm_load_2d(const void* gaddr, unsigned ldsByteOff,
                                            unsigned widthUnits, unsigned rows,
                                            unsigned strideUnits) {
  const unsigned long long ga = (unsigned long long)gaddr;
  u32x4 g0;
  g0[0] = 1u;                                            // count=1, user mode
  g0[1] = ldsByteOff;                                    // lds_addr [63:32]
  g0[2] = (unsigned)(ga & 0xFFFFFFFFu);                  // global_addr lo
  g0[3] = (unsigned)((ga >> 32) & 0x1FFFFFFu) | (2u << 30); // addr hi | type=2
  const unsigned dim0 = strideUnits;                     // tensor_dim0 (>= tile)
  const unsigned dim1 = rows;
  i32x8 g1;
  g1[0] = (int)(3u << 16);                               // data_size = 8B
  g1[1] = (int)((dim0 & 0xFFFFu) << 16);                 // [63:48] dim0 lo16
  g1[2] = (int)(((dim0 >> 16) & 0xFFFFu) | ((dim1 & 0xFFFFu) << 16));
  g1[3] = (int)(((dim1 >> 16) & 0xFFFFu) | ((widthUnits & 0xFFFFu) << 16)); // tile_dim0
  g1[4] = (int)(rows & 0xFFFFu);                         // tile_dim1 (tile_dim2=0)
  g1[5] = (int)strideUnits;                              // dim0_stride lo32
  g1[6] = 0;                                             // dim0_stride hi16
  g1[7] = 0;
  i32x4 z4 = { 0, 0, 0, 0 };
  i32x8 z8 = { 0, 0, 0, 0, 0, 0, 0, 0 };
  // clang-23 / therock arity: (g0, g1, g2, g3, extra, cpol)
  __builtin_amdgcn_tensor_load_to_lds(g0, g1, z4, z4, z8, 0);
}
#else
#define HAVE_TDM 0
#endif

// ---------------------------------------------------------------------------
// Prep kernels
// ---------------------------------------------------------------------------
__global__ void cvt_f32_f16(_Float16* dst, const float* src, int n) {
  int i = blockIdx.x * 256 + threadIdx.x;
  if (i < n) dst[i] = (_Float16)src[i];
}
__global__ void build_wcat(_Float16* wcat, const float* w_ih, const float* w_hh) {
  int i = blockIdx.x * 256 + threadIdx.x;          // 2 * 2048 * 1024
  int l  = i >> 21;
  int r  = i & ((1 << 21) - 1);
  int n  = r >> 10;
  int kd = r & 1023;
  float v = (kd < HID) ? w_ih[(l * G4 + n) * HID + kd]
                       : w_hh[(l * G4 + n) * HID + (kd - HID)];
  wcat[i] = (_Float16)v;
}
__global__ void build_bsum(float* bsum, const float* b_ih, const float* b_hh) {
  int i = blockIdx.x * 256 + threadIdx.x;          // 2 * 2048
  if (i < 2 * G4) bsum[i] = b_ih[i] + b_hh[i];
}
__global__ void zero_f16(_Float16* p, int n) {
  int i = blockIdx.x * 256 + threadIdx.x; if (i < n) p[i] = (_Float16)0.f;
}
__global__ void zero_f32(float* p, int n) {
  int i = blockIdx.x * 256 + threadIdx.x; if (i < n) p[i] = 0.f;
}

// ---------------------------------------------------------------------------
// QKV projection: q/k stored (n,h,t,hd) f16 ; v stored transposed (n,h,hd,t).
// ---------------------------------------------------------------------------
__global__ void qkv_proj(const float* __restrict__ x,
                         const _Float16* __restrict__ wq16,
                         const _Float16* __restrict__ wk16,
                         const _Float16* __restrict__ wv16,
                         _Float16* __restrict__ q, _Float16* __restrict__ k,
                         _Float16* __restrict__ vT) {
  const int n  = blockIdx.x >> 4;
  const int tt = blockIdx.x & 15;
  const int h  = threadIdx.x >> 5;

  __shared__ _Float16 stg[8][256];

  v16h a = ldA32cvt(x + ((size_t)(n * TT + tt * 16)) * FEATD + h * HD, FEATD);

  const size_t rowbase = ((size_t)(n * NHEAD + h) * TT + tt * 16) * HD;
#pragma unroll
  for (int i0 = 0; i0 < 32; i0 += 16) {
    v8f z = {};
    v8f cq = WMMA(a, ldB16(wq16 + i0 * HD, HD), z);
    v8f ck = WMMA(a, ldB16(wk16 + i0 * HD, HD), z);
    v8f cv = WMMA(a, ldB16(wv16 + i0 * HD, HD), z);
    stC16co(q + rowbase + i0, HD, cq, stg[h]);
    stC16co(k + rowbase + i0, HD, ck, stg[h]);
    stCtr16co(vT + ((size_t)(n * NHEAD + h) * HD + i0) * TT + tt * 16, TT, cv, stg[h]);
  }
}

// ---------------------------------------------------------------------------
// Attention: one block per (n,h); 8 waves; wave w does q-tiles {w, w+8}.
// ---------------------------------------------------------------------------
__global__ void attention(const _Float16* __restrict__ q,
                          const _Float16* __restrict__ k,
                          const _Float16* __restrict__ vT,
                          _Float16* __restrict__ xa) {
  const int n = blockIdx.x >> 3;
  const int h = blockIdx.x & 7;
  const int w = threadIdx.x >> 5;
  const int l = laneid();
  const int col = l & 15;
  const int rb  = (l & 16) ? 8 : 0;

  __shared__ _Float16 kLds[TT * HD];       // 16 KB
  __shared__ _Float16 vLds[HD * TT];       // 16 KB
  __shared__ _Float16 pst[8][16 * 32];     //  8 KB

  const _Float16* qb  = q  + (size_t)(n * NHEAD + h) * TT * HD;
  const _Float16* kb  = k  + (size_t)(n * NHEAD + h) * TT * HD;
  const _Float16* vtb = vT + (size_t)(n * NHEAD + h) * HD * TT;

  for (int i = threadIdx.x; i < TT * HD / 8; i += 256) {
    ((v8h*)kLds)[i] = ((const v8h*)kb)[i];
    ((v8h*)vLds)[i] = ((const v8h*)vtb)[i];
  }
  __syncthreads();

  for (int it = 0; it < 2; ++it) {
    const int qt = w + it * 8;
    v16h aq = ldA16(qb + qt * 16 * HD, HD);

    v8f s[16];
#pragma unroll
    for (int kt = 0; kt < 16; ++kt) {
      v8f z = {};
      s[kt] = WMMA(aq, ldB16(kLds + kt * 16 * HD, HD), z);
    }

    float rmax[8], rsum[8];
#pragma unroll
    for (int v = 0; v < 8; ++v) {
      float m = s[0][v];
#pragma unroll
      for (int kt = 1; kt < 16; ++kt) m = fmaxf(m, s[kt][v]);
      m = fmaxf(m, __shfl_xor(m, 1));
      m = fmaxf(m, __shfl_xor(m, 2));
      m = fmaxf(m, __shfl_xor(m, 4));
      m = fmaxf(m, __shfl_xor(m, 8));
      rmax[v] = m; rsum[v] = 0.f;
    }
    const float c0 = 0.0625f * 1.44269504f;
#pragma unroll
    for (int kt = 0; kt < 16; ++kt)
#pragma unroll
      for (int v = 0; v < 8; ++v) {
        float e = exp2f((s[kt][v] - rmax[v]) * c0);
        s[kt][v] = e; rsum[v] += e;
      }
#pragma unroll
    for (int v = 0; v < 8; ++v) {
      float r = rsum[v];
      r += __shfl_xor(r, 1); r += __shfl_xor(r, 2);
      r += __shfl_xor(r, 4); r += __shfl_xor(r, 8);
      rsum[v] = r;
    }

    v8f o0 = {}, o1 = {};
    for (int ch = 0; ch < 8; ++ch) {
#pragma unroll
      for (int v = 0; v < 8; ++v) {
        pst[w][(rb + v) * 32 + col]      = (_Float16)s[2 * ch][v];
        pst[w][(rb + v) * 32 + 16 + col] = (_Float16)s[2 * ch + 1][v];
      }
      v16h ap = ldA16(&pst[w][0], 32);
      o0 = WMMA(ap, ldB16(vLds + 0 * 16 * TT + ch * 32, TT), o0);
      o1 = WMMA(ap, ldB16(vLds + 1 * 16 * TT + ch * 32, TT), o1);
    }
#pragma unroll
    for (int v = 0; v < 8; ++v) { o0[v] /= rsum[v]; o1[v] /= rsum[v]; }

    _Float16* ob = xa + ((size_t)(n * TT + qt * 16)) * FEATD + h * HD;
    stC16co(ob,      FEATD, o0, &pst[w][0]);
    stC16co(ob + 16, FEATD, o1, &pst[w][0]);
  }
}

// ---------------------------------------------------------------------------
// Generic WMMA GEMM:  out = A(f16, MxK) @ W(f16, (N,K)).T + bias
// ---------------------------------------------------------------------------
__global__ void gemm_wmma(const _Float16* __restrict__ A, int lda,
                          const _Float16* __restrict__ W, int ldw,
                          const float* __restrict__ bias,
                          _Float16* __restrict__ outH, int ldo,
                          int Ntiles, int Kchunks) {
  const int w = threadIdx.x >> 5;
  const int tile0 = blockIdx.x * 8;
  const int mt = tile0 / Ntiles;
  const int nt = tile0 % Ntiles + w;
  const int K  = Kchunks * 32;

  __shared__ _Float16 aLds[16 * 256];   // 8 KB (K <= 256)
  __shared__ _Float16 stg[8][256];      // 4 KB

  for (int i = threadIdx.x; i < 2 * K; i += 256) {
    const int row = (i * 8) / K;
    const int off = (i * 8) % K;
    ((v8h*)aLds)[i] = *(const v8h*)(A + (size_t)(mt * 16 + row) * lda + off);
  }
  __syncthreads();

  const _Float16* wb = W + (size_t)nt * 16 * ldw;
  v8f c = {};
  for (int kk = 0; kk < Kchunks; ++kk) {
    if (kk + 1 < Kchunks) __builtin_prefetch(wb + (kk + 1) * 32, 0, 0);
    v16h a = ldA16(aLds + kk * 32, K);
    v16h b = ldB16(wb + kk * 32, ldw);
    c = WMMA(a, b, c);
  }
  const float bv = bias[nt * 16 + (laneid() & 15)];
#pragma unroll
  for (int v = 0; v < 8; ++v) c[v] += bv;
  stC16co(outH + (size_t)mt * 16 * ldo + nt * 16, ldo, c, stg[w]);
}

// ---------------------------------------------------------------------------
// LSTM gates GEMM: gates(256 x 2048) = [X | Hprev](256 x 1024) @ Wcat.T + bsum
// A-tile staged via TDM (tensor_load_to_lds) when available: two 2-D tile
// DMAs (X half, H half) issued by wave 0, s_wait_tensorcnt, barrier.
// LDS carved from one shared block so D#.lds_addr offsets are known.
// ---------------------------------------------------------------------------
__global__ void lstm_gates(const _Float16* __restrict__ X, int ldx,
                           const _Float16* __restrict__ Hprev,
                           const _Float16* __restrict__ Wcat,
                           const float* __restrict__ bsum,
                           float* __restrict__ gates) {
  const int w = threadIdx.x >> 5;
  const int tile0 = blockIdx.x * 8;
  const int mt = tile0 >> 7;          // 16 M-tiles
  const int nt = (tile0 & 127) + w;   // 128 N-tiles

  // [0,16384): X half (16x512 f16); [16384,32768): H half; then C scratch.
  __shared__ __align__(16) char smem[40960];
  _Float16* aX   = (_Float16*)smem;
  _Float16* aH   = (_Float16*)(smem + 16384);
  float*    stgf = (float*)(smem + 32768);

#if HAVE_TDM
  if ((threadIdx.x >> 5) == 0) {        // wave-uniform: wave 0 drives the TDM
    tdm_load_2d(X + (size_t)mt * 16 * ldx,      0u, 128u, 16u,
                (unsigned)((size_t)ldx * 2 / 8));
    tdm_load_2d(Hprev + (size_t)mt * 16 * HID, 16384u, 128u, 16u, 128u);
    __builtin_amdgcn_s_wait_tensorcnt(0);
  }
  __syncthreads();
#else
  for (int i = threadIdx.x; i < 1024; i += 256) {     // X half: 16*512/8
    const int row = i >> 6, off = (i & 63) * 8;
    ((v8h*)aX)[i] = *(const v8h*)(X + (size_t)(mt * 16 + row) * ldx + off);
  }
  for (int i = threadIdx.x; i < 1024; i += 256) {     // H half
    const int row = i >> 6, off = (i & 63) * 8;
    ((v8h*)aH)[i] = *(const v8h*)(Hprev + (size_t)(mt * 16 + row) * HID + off);
  }
  __syncthreads();
#endif

  const _Float16* wb = Wcat + (size_t)nt * 16 * 1024;
  v8f c = {};
  for (int kk = 0; kk < 32; ++kk) {
    __builtin_prefetch(wb + ((kk + 1) & 31) * 32, 0, 0);
    v16h a = (kk < 16) ? ldA16(aX + kk * 32, HID)
                       : ldA16(aH + (kk - 16) * 32, HID);
    v16h b = ldB16(wb + kk * 32, 1024);
    c = WMMA(a, b, c);
  }
  const float bv = bsum[nt * 16 + (laneid() & 15)];
#pragma unroll
  for (int v = 0; v < 8; ++v) c[v] += bv;
  stC32co(gates + (size_t)mt * 16 * G4 + nt * 16, G4, c, stgf + w * 256);
}

// ---------------------------------------------------------------------------
// LSTM cell elementwise (PyTorch gate order i,f,g,o).
// ---------------------------------------------------------------------------
__device__ __forceinline__ float sigf(float x) { return 1.f / (1.f + expf(-x)); }

__global__ void lstm_cell(const float* __restrict__ gates,
                          float* __restrict__ cst, _Float16* __restrict__ hst) {
  const int idx = blockIdx.x * 256 + threadIdx.x;   // 256*512
  const int m = idx >> 9;
  const int j = idx & 511;
  const float gi = gates[m * G4 + j];
  const float gf = gates[m * G4 + HID + j];
  const float gg = gates[m * G4 + 2 * HID + j];
  const float go = gates[m * G4 + 3 * HID + j];
  const float cn = sigf(gf) * cst[idx] + sigf(gi) * tanhf(gg);
  cst[idx] = cn;
  hst[idx] = (_Float16)(sigf(go) * tanhf(cn));
}

// ---------------------------------------------------------------------------
// Output head: ypred[n, t, qh] = b_out[qh] + sum_j relu(h1[n,j]) * w_out[qh,j]
// ---------------------------------------------------------------------------
__global__ void out_proj(const _Float16* __restrict__ h1,
                         const float* __restrict__ w_out,
                         const float* __restrict__ b_out,
                         float* __restrict__ y, int t) {
  const int wave = (blockIdx.x * 256 + threadIdx.x) >> 5;
  const int n  = wave >> 3;
  const int qh = wave & 7;
  const int l  = laneid();
  float s = 0.f;
#pragma unroll
  for (int jj = 0; jj < 16; ++jj) {
    const int j = l + jj * 32;
    s += fmaxf((float)h1[n * HID + j], 0.f) * w_out[qh * HID + j];
  }
  s += __shfl_xor(s, 1);  s += __shfl_xor(s, 2);
  s += __shfl_xor(s, 4);  s += __shfl_xor(s, 8);  s += __shfl_xor(s, 16);
  if (l == 0) y[((size_t)n * (TT - 1) + t) * NHEAD + qh] = s + b_out[qh];
}

// ---------------------------------------------------------------------------
// Host launcher
// ---------------------------------------------------------------------------
extern "C" void kernel_launch(void* const* d_in, const int* in_sizes, int n_in,
                              void* d_out, int out_size, void* d_ws, size_t ws_size,
                              hipStream_t stream) {
  const float* x     = (const float*)d_in[0];
  const float* wq    = (const float*)d_in[1];
  const float* wk    = (const float*)d_in[2];
  const float* wv    = (const float*)d_in[3];
  const float* w_fc  = (const float*)d_in[4];
  const float* b_fc  = (const float*)d_in[5];
  const float* w_hid = (const float*)d_in[6];
  const float* b_hid = (const float*)d_in[7];
  const float* w_ih  = (const float*)d_in[8];
  const float* w_hh  = (const float*)d_in[9];
  const float* b_ih  = (const float*)d_in[10];
  const float* b_hh  = (const float*)d_in[11];
  const float* w_out = (const float*)d_in[12];
  const float* b_out = (const float*)d_in[13];
  float* out = (float*)d_out;

  char* ws = (char*)d_ws;
  size_t cur = 0;
  auto alloc = [&](size_t bytes) { size_t o = cur; cur += (bytes + 255) & ~(size_t)255; return o; };

  const size_t qkvB = (size_t)BB * NHEAD * TT * HD * 2;   // 32 MB each
  _Float16* qbuf  = (_Float16*)(ws + alloc(qkvB));
  _Float16* kbuf  = (_Float16*)(ws + alloc(qkvB));
  _Float16* vTbuf = (_Float16*)(ws + alloc(qkvB));
  _Float16* xabuf = (_Float16*)(ws + alloc((size_t)BB * TT * FEATD * 2));
  _Float16* xcbuf = (_Float16*)(ws + alloc((size_t)BB * TT * HID * 2));
  _Float16* wq16  = (_Float16*)(ws + alloc(HD * HD * 2));
  _Float16* wk16  = (_Float16*)(ws + alloc(HD * HD * 2));
  _Float16* wv16  = (_Float16*)(ws + alloc(HD * HD * 2));
  _Float16* wfc16 = (_Float16*)(ws + alloc((size_t)FEATD * FEATD * 2));
  _Float16* whid16= (_Float16*)(ws + alloc((size_t)HID * FEATD * 2));
  _Float16* wcat  = (_Float16*)(ws + alloc((size_t)2 * G4 * 1024 * 2));
  float*    bsum  = (float*)(ws + alloc((size_t)2 * G4 * 4));
  _Float16* hbuf  = (_Float16*)(ws + alloc((size_t)2 * BB * HID * 2)); // h[2]
  float*    cbuf  = (float*)(ws + alloc((size_t)2 * BB * HID * 4));    // c[2]
  float*    gates = (float*)(ws + alloc((size_t)BB * G4 * 4));
  _Float16* y1buf = qbuf;   // reuse: q dead after attention, y1 born after

  // ---- prep ----
  cvt_f32_f16<<<4, 256, 0, stream>>>(wq16, wq, HD * HD);
  cvt_f32_f16<<<4, 256, 0, stream>>>(wk16, wk, HD * HD);
  cvt_f32_f16<<<4, 256, 0, stream>>>(wv16, wv, HD * HD);
  cvt_f32_f16<<<FEATD * FEATD / 256, 256, 0, stream>>>(wfc16, w_fc, FEATD * FEATD);
  cvt_f32_f16<<<HID * FEATD / 256, 256, 0, stream>>>(whid16, w_hid, HID * FEATD);
  build_wcat<<<2 * G4 * 1024 / 256, 256, 0, stream>>>(wcat, w_ih, w_hh);
  build_bsum<<<16, 256, 0, stream>>>(bsum, b_ih, b_hh);
  zero_f16<<<2 * BB * HID / 256, 256, 0, stream>>>(hbuf, 2 * BB * HID);
  zero_f32<<<2 * BB * HID / 256, 256, 0, stream>>>(cbuf, 2 * BB * HID);

  // ---- attention ----
  qkv_proj<<<BB * (TT / 16), 256, 0, stream>>>(x, wq16, wk16, wv16, qbuf, kbuf, vTbuf);
  attention<<<BB * NHEAD, 256, 0, stream>>>(qbuf, kbuf, vTbuf, xabuf);

  // ---- FC layers ----
  gemm_wmma<<<(BB * TT / 16) * (FEATD / 16) / 8, 256, 0, stream>>>(
      xabuf, FEATD, wfc16, FEATD, b_fc, y1buf, FEATD, FEATD / 16, FEATD / 32);
  gemm_wmma<<<(BB * TT / 16) * (HID / 16) / 8, 256, 0, stream>>>(
      y1buf, FEATD, whid16, FEATD, b_hid, xcbuf, HID, HID / 16, FEATD / 32);

  // ---- sequential LSTM ----
  _Float16* h0 = hbuf;
  _Float16* h1 = hbuf + BB * HID;
  float* c0 = cbuf;
  float* c1 = cbuf + BB * HID;
  for (int t = 0; t < TT; ++t) {
    lstm_gates<<<256, 256, 0, stream>>>(xcbuf + (size_t)t * HID, TT * HID,
                                        h0, wcat, bsum, gates);
    lstm_cell<<<BB * HID / 256, 256, 0, stream>>>(gates, c0, h0);
    lstm_gates<<<256, 256, 0, stream>>>(h0, HID,
                                        h1, wcat + (size_t)G4 * 1024, bsum + G4, gates);
    lstm_cell<<<BB * HID / 256, 256, 0, stream>>>(gates, c1, h1);
    if (t < TT - 1)
      out_proj<<<256, 256, 0, stream>>>(h1, w_out, b_out, out, t);
  }
}
